// Model_31559419691320
// MI455X (gfx1250) — matrix operand order repaired and checked by reading, not compiled, not run on hardware
//
#include <hip/hip_runtime.h>

#define BSZ 64
#define NTOK 512
#define NENT 64
#define DIM 768
#define RDIM 7

typedef __attribute__((ext_vector_type(16))) __bf16 v16bf;
typedef __attribute__((ext_vector_type(8)))  __bf16 v8bf;
typedef __attribute__((ext_vector_type(8)))  float  v8f;

// ---------------------------------------------------------------------------
// Kernel 1: ragged mean-pool over token spans, emit bf16 entity encodings.
// One block per entity (4096 blocks), 256 threads -> 3 dims each.
// ---------------------------------------------------------------------------
__global__ __launch_bounds__(256) void pool_kernel(
    const float* __restrict__ bert, const int* __restrict__ starts,
    const int* __restrict__ ends, __bf16* __restrict__ ent) {
  int e = blockIdx.x;           // flat [b, ne]
  int b = e >> 6;
  int u = starts[e];
  int v = ends[e];
  int len = v - u; if (len < 1) len = 1;
  float inv = 1.0f / (float)len;
  int d0 = threadIdx.x * 3;
  const float* base = bert + ((size_t)b * NTOK) * DIM + d0;
  float s0 = 0.f, s1 = 0.f, s2 = 0.f;
  for (int t = u; t < v; ++t) {
    const float* row = base + (size_t)t * DIM;
    s0 += row[0]; s1 += row[1]; s2 += row[2];
  }
  __bf16* o = ent + (size_t)e * DIM + d0;
  o[0] = (__bf16)(s0 * inv);
  o[1] = (__bf16)(s1 * inv);
  o[2] = (__bf16)(s2 * inv);
}

// ---------------------------------------------------------------------------
// Kernel 0b: f32 -> bf16 convert (used for the weight matrices).
// ---------------------------------------------------------------------------
__global__ __launch_bounds__(256) void cvt_kernel(
    const float* __restrict__ in, __bf16* __restrict__ out, int n) {
  int i = blockIdx.x * 256 + threadIdx.x;
  if (i < n) out[i] = (__bf16)in[i];
}

// ---------------------------------------------------------------------------
// Kernel 2: Y = X @ W^T + b, bf16 in / bf16 out, f32 WMMA accumulation.
// M=4096, N=768, K=768. One 16x16 C-tile per wave; 8 waves per block.
// A-frag: lane<16 -> row M=lane, K {0..7,16..23}; lane>=16 -> K {8..15,24..31}.
// B-frag: column N=lane%16 (= row of W), same K split as A-frag.
// ---------------------------------------------------------------------------
__global__ __launch_bounds__(256) void proj_kernel(
    const __bf16* __restrict__ X, const __bf16* __restrict__ Wb,
    const float* __restrict__ bias, __bf16* __restrict__ Y) {
  int wave = threadIdx.x >> 5;
  int lane = threadIdx.x & 31;
  int gid  = blockIdx.x * 8 + wave;        // 0..12287
  int mtile = gid / 48;
  int ntile = gid % 48;
  int mbase = mtile * 16, nbase = ntile * 16;
  int lh = lane >> 4;                      // half-wave select
  int lm = lane & 15;

  const __bf16* arow = X  + (size_t)(mbase + lm) * DIM + lh * 8;
  const __bf16* brow = Wb + (size_t)(nbase + lm) * DIM + lh * 8;

  v8f acc = {0.f, 0.f, 0.f, 0.f, 0.f, 0.f, 0.f, 0.f};
  for (int kc = 0; kc < DIM; kc += 32) {
    v8bf alo = *(const v8bf*)(arow + kc);
    v8bf ahi = *(const v8bf*)(arow + kc + 16);
    v16bf a;
#pragma unroll
    for (int i = 0; i < 8; ++i) { a[i] = alo[i]; a[8 + i] = ahi[i]; }
    v8bf blo = *(const v8bf*)(brow + kc);
    v8bf bhi = *(const v8bf*)(brow + kc + 16);
    v16bf bfrag;
#pragma unroll
    for (int i = 0; i < 8; ++i) { bfrag[i] = blo[i]; bfrag[8 + i] = bhi[i]; }
    acc = __builtin_amdgcn_wmma_f32_16x16x32_bf16(
        false, a, false, bfrag, (short)0, acc, false, false);
  }

  float bv = bias[nbase + lm];
#pragma unroll
  for (int vi = 0; vi < 8; ++vi) {
    int m = mbase + lh * 8 + vi;           // C layout: lanes 16-31 hold M=v+8
    Y[(size_t)m * DIM + nbase + lm] = (__bf16)(acc[vi] + bv);
  }
}

// ---------------------------------------------------------------------------
// Kernel 3: alpha[b,i,j,r] = sum_d q[b,i,d]*k[b,j,d]*ln_w[r,d] + ln_b[r],
// masked. Per wave: one 16x16 (i,j) tile, 7 f32 accumulators (one per r).
// ln_w staged in LDS as bf16; scale folded into the A-frag with packed
// bf16 multiplies (v_pk_mul_bf16) so the VALU work co-executes with WMMA.
// ---------------------------------------------------------------------------
__global__ __launch_bounds__(256) void alpha_kernel(
    const __bf16* __restrict__ Q, const __bf16* __restrict__ Km,
    const float* __restrict__ lnw, const float* __restrict__ lnb,
    const float* __restrict__ mask, float* __restrict__ out) {
  __shared__ __bf16 lw[RDIM * DIM];
  for (int i = threadIdx.x; i < RDIM * DIM; i += 256)
    lw[i] = (__bf16)lnw[i];
  __syncthreads();

  int wave = threadIdx.x >> 5;
  int lane = threadIdx.x & 31;
  int gid  = blockIdx.x * 8 + wave;        // 0..1023
  int b    = gid >> 4;
  int pair = gid & 15;
  int ibase = (pair >> 2) * 16;
  int jbase = (pair & 3) * 16;
  int lh = lane >> 4;
  int lm = lane & 15;

  const __bf16* qrow = Q  + (size_t)(b * NENT + ibase + lm) * DIM + lh * 8;
  const __bf16* krow = Km + (size_t)(b * NENT + jbase + lm) * DIM + lh * 8;

  v8f acc[RDIM];
#pragma unroll
  for (int r = 0; r < RDIM; ++r)
    acc[r] = (v8f){0.f, 0.f, 0.f, 0.f, 0.f, 0.f, 0.f, 0.f};

  for (int kc = 0; kc < DIM; kc += 32) {
    v8bf qlo = *(const v8bf*)(qrow + kc);
    v8bf qhi = *(const v8bf*)(qrow + kc + 16);
    v8bf klo = *(const v8bf*)(krow + kc);
    v8bf khi = *(const v8bf*)(krow + kc + 16);
    v16bf kb;
#pragma unroll
    for (int i = 0; i < 8; ++i) { kb[i] = klo[i]; kb[8 + i] = khi[i]; }
#pragma unroll
    for (int r = 0; r < RDIM; ++r) {
      const __bf16* lp = lw + r * DIM + kc + lh * 8;
      v8bf wlo = *(const v8bf*)(lp);        // ds_load_b128
      v8bf whi = *(const v8bf*)(lp + 16);
      v8bf plo = qlo * wlo;                 // packed bf16 mul
      v8bf phi = qhi * whi;
      v16bf a;
#pragma unroll
      for (int i = 0; i < 8; ++i) { a[i] = plo[i]; a[8 + i] = phi[i]; }
      acc[r] = __builtin_amdgcn_wmma_f32_16x16x32_bf16(
          false, a, false, kb, (short)0, acc[r], false, false);
    }
  }

#pragma unroll
  for (int vi = 0; vi < 8; ++vi) {
    int i = ibase + lh * 8 + vi;
    int j = jbase + lm;
    float mm = mask[b * NENT + i] * mask[b * NENT + j];
    float* op = out + ((size_t)(b * NENT + i) * NENT + j) * RDIM;
#pragma unroll
    for (int r = 0; r < RDIM; ++r) op[r] = (acc[r][vi] + lnb[r]) * mm;
  }
}

// ---------------------------------------------------------------------------
// Launcher. Workspace layout (bytes):
//   [0)              ent_bf16  4096*768*2 = 6,291,456
//   [+6291456)       wq_bf16    768*768*2 = 1,179,648
//   [+7471104)       wk_bf16              = 1,179,648
//   [+8650752)       q_bf16               = 6,291,456
//   [+14942208)      k_bf16               = 6,291,456   (total ~21.2 MB)
// ---------------------------------------------------------------------------
extern "C" void kernel_launch(void* const* d_in, const int* in_sizes, int n_in,
                              void* d_out, int out_size, void* d_ws,
                              size_t ws_size, hipStream_t stream) {
  const float* bert   = (const float*)d_in[0];
  const float* emask  = (const float*)d_in[1];
  const float* wq_w   = (const float*)d_in[2];
  const float* wq_b   = (const float*)d_in[3];
  const float* wk_w   = (const float*)d_in[4];
  const float* wk_b   = (const float*)d_in[5];
  const float* ln_w   = (const float*)d_in[6];
  const float* ln_b   = (const float*)d_in[7];
  const int*   starts = (const int*)d_in[8];
  const int*   ends   = (const int*)d_in[9];

  char* ws = (char*)d_ws;
  __bf16* ent = (__bf16*)(ws);
  __bf16* wqb = (__bf16*)(ws + 6291456);
  __bf16* wkb = (__bf16*)(ws + 6291456 + 1179648);
  __bf16* qb  = (__bf16*)(ws + 6291456 + 2 * 1179648);
  __bf16* kb  = (__bf16*)(ws + 2 * 6291456 + 2 * 1179648);

  const int WELEMS = DIM * DIM;  // 589824

  pool_kernel<<<BSZ * NENT, 256, 0, stream>>>(bert, starts, ends, ent);
  cvt_kernel<<<(WELEMS + 255) / 256, 256, 0, stream>>>(wq_w, wqb, WELEMS);
  cvt_kernel<<<(WELEMS + 255) / 256, 256, 0, stream>>>(wk_w, wkb, WELEMS);

  // 4096/16 * 768/16 = 12288 waves, 8 waves/block -> 1536 blocks
  proj_kernel<<<1536, 256, 0, stream>>>(ent, wqb, wq_b, qb);
  proj_kernel<<<1536, 256, 0, stream>>>(ent, wkb, wk_b, kb);

  // 64 batches * 16 (i,j) tile pairs = 1024 waves -> 128 blocks
  alpha_kernel<<<128, 256, 0, stream>>>(qb, kb, ln_w, ln_b, emask,
                                        (float*)d_out);
}
